// Model_70970039599123
// MI455X (gfx1250) — compile-verified, matrix operand order checked by reading, not run on hardware
//
#include <hip/hip_runtime.h>
#include <math.h>
#include <stdint.h>

// Problem dims (compile-time, from the reference)
#define B_DIM   32
#define S_DIM   512
#define EMB_DIM 512
#define H_DIM   1024
#define G_DIM   4096   // 4*H
#define V_DIM   8192
#define MROWS   (B_DIM * S_DIM)   // 16384 rows in [S,B] flattening (row = s*32+b)

typedef __attribute__((ext_vector_type(16))) __bf16        v16bf;
typedef __attribute__((ext_vector_type(8)))  float         f32x8;
typedef __attribute__((ext_vector_type(4)))  unsigned int  u32x4;
typedef __attribute__((ext_vector_type(8)))  int           i32x8;
typedef __attribute__((ext_vector_type(4)))  int           i32x4;

#if __has_builtin(__builtin_amdgcn_tensor_load_to_lds)
#define HAVE_TDM 1
#else
#define HAVE_TDM 0
#endif

union Frag32B {        // one WMMA 16-bit operand fragment: 16 bf16 per lane = 32 bytes
    v16bf v;
    uint4 u[2];
};

__device__ __forceinline__ unsigned short f2bf(float f) {
    unsigned int u = __float_as_uint(f);
    u += 0x7FFFu + ((u >> 16) & 1u);   // round-to-nearest-even
    return (unsigned short)(u >> 16);
}

__device__ __forceinline__ float sigm(float x) { return 1.0f / (1.0f + __expf(-x)); }

__device__ __forceinline__ f32x8 wmma_bf16(const Frag32B& a, const Frag32B& b, f32x8 c) {
    return __builtin_amdgcn_wmma_f32_16x16x32_bf16(false, a.v, false, b.v,
                                                   (short)0, c, false, false);
}

// async DMA: global -> LDS, 16 bytes per lane, tracked by ASYNCcnt
__device__ __forceinline__ void async_b128(unsigned lds_off, const void* gaddr) {
    asm volatile("global_load_async_to_lds_b128 %0, %1, off"
                 :: "v"(lds_off), "v"(gaddr) : "memory");
}

__device__ __forceinline__ void wait_async0() {
    asm volatile("s_wait_asynccnt 0" ::: "memory");
}

__device__ __forceinline__ void wait_tensor0() {
#if __has_builtin(__builtin_amdgcn_s_wait_tensorcnt)
    __builtin_amdgcn_s_wait_tensorcnt(0);
#else
    asm volatile("s_wait_tensorcnt 0" ::: "memory");
#endif
}

#if HAVE_TDM
// TDM: one instruction DMAs a 2D tile (tile_w x tile_h, 2-byte elems) from
// global (row stride = stride_elems) into LDS with a padded pitch of
// tile_w*2 + 16 bytes (pad_interval = 32 DWORDs = one 128B row,
// pad_amount = 4 DWORDs = 16B).  D# layout per CDNA5 ISA ch.8.
// This toolchain exposes the 6-arg builtin:
//   (u32x4 g0, i32x8 g1, i32x4 g2, i32x4 g3, i32x8 g4, i32 cpol)
__device__ __forceinline__ void tdm_load_2d(unsigned lds_addr, const void* gaddr,
                                            int tile_w, int tile_h, int stride_elems) {
    const uint64_t ga = (uint64_t)(uintptr_t)gaddr;
    u32x4 g0;
    g0[0] = 1u;                                             // count=1, user D#
    g0[1] = lds_addr;                                       // lds_addr[31:0]
    g0[2] = (unsigned)(ga & 0xFFFFFFFFu);                   // global_addr lo
    g0[3] = (unsigned)((ga >> 32) & 0x01FFFFFFu) | (2u << 30);  // addr hi | type=2
    i32x8 g1;
    g1[0] = (1 << 16)          // data_size = 2 bytes
          | (1 << 20)          // pad_enable
          | (4 << 22)          // pad_interval: 32 DWORDs (128B row)
          | (3 << 25);         // pad_amount: 4 DWORDs (16B) -> 144B pitch
    g1[1] = (tile_w & 0xFFFF) << 16;                        // tensor_dim0 = tile_w
    g1[2] = ((tile_w >> 16) & 0xFFFF) | ((tile_h & 0xFFFF) << 16); // tensor_dim1 = tile_h
    g1[3] = ((tile_h >> 16) & 0xFFFF) | ((tile_w & 0xFFFF) << 16); // tile_dim0
    g1[4] = (tile_h & 0xFFFF);                              // tile_dim1 (tile_dim2=0)
    g1[5] = stride_elems;                                   // tensor_dim0_stride lo
    g1[6] = 0;
    g1[7] = 0;
    const i32x4 zz4 = {0, 0, 0, 0};
    const i32x8 zz8 = {0, 0, 0, 0, 0, 0, 0, 0};
    __builtin_amdgcn_tensor_load_to_lds(g0, g1, zz4, zz4, zz8, 0);
}
#endif

// ---------------------------------------------------------------------------
// fp32 -> bf16 bulk convert
// ---------------------------------------------------------------------------
__global__ void cvt_bf16_kernel(const float* __restrict__ in,
                                unsigned short* __restrict__ out, size_t n) {
    size_t i = (size_t)blockIdx.x * blockDim.x + threadIdx.x;
    size_t stride = (size_t)gridDim.x * blockDim.x;
    for (; i < n; i += stride) out[i] = f2bf(in[i]);
}

// ---------------------------------------------------------------------------
// Embedding gather: xs[s*32+b][e] = bf16(emb_W[e][x[b][s]] + emb_b[e])
// ---------------------------------------------------------------------------
__global__ void embed_kernel(const int* __restrict__ x,
                             const float* __restrict__ embW,
                             const float* __restrict__ embB,
                             unsigned short* __restrict__ xs) {
    const int sb = blockIdx.x;            // sb = s*32 + b
    const int s = sb >> 5, b = sb & 31;
    const int tok = x[b * S_DIM + s];
    for (int e = threadIdx.x; e < EMB_DIM; e += blockDim.x) {
        float v = embW[(size_t)e * V_DIM + tok] + embB[e];
        xs[(size_t)sb * EMB_DIM + e] = f2bf(v);
    }
}

// ---------------------------------------------------------------------------
// TDM + async double-buffered NT GEMM:
//   C[M,N] (f32) = A[M,K](bf16) * B[N,K]^T(bf16) + bias0 + bias1
// Block = 256 thr = 8 waves (2 along M x 4 along N), wave tile 32x32 (4 WMMA acc).
// Block tile 64(M) x 128(N); K-slab = 64, double buffered in LDS:
//   - A slab (64x64, 8KB):  ASYNCcnt path, 2x global_load_async_to_lds_b128/thr
//   - B slab (128x64,16KB): TENSORcnt path, ONE tensor_load_to_lds from wave 0
//     (TDM pad fields reproduce the 144B padded pitch)
// Pipeline: issue(k+1) -> compute(k) -> s_wait_asynccnt/tensorcnt -> barrier.
// LDS rows padded to 72 bf16 (144B) for conflict-free b128 fragment reads.
// remapBS: out row = (m%32)*S + m/32 ([s,b] -> [b,s] classifier layout).
// Grids divide exactly; no bounds guards (keeps EXEC/barriers uniform).
// ---------------------------------------------------------------------------
#define LDSPITCH 72
#define SLABK    64

__global__ __launch_bounds__(256)
void gemm_bf16_wmma_async(const unsigned short* __restrict__ A, int lda,
                          const unsigned short* __restrict__ Bw, int ldb,
                          float* __restrict__ C, int ldc,
                          const float* __restrict__ bias0,
                          const float* __restrict__ bias1,
                          int M, int N, int K, int remapBS) {
    __shared__ unsigned short shA[2][64][LDSPITCH];
    __shared__ unsigned short shB[2][128][LDSPITCH];

    const int tid  = threadIdx.x;
    const int lane = tid & 31;
    const int wave = tid >> 5;
    const int wm = wave & 1;             // 2 waves along M
    const int wn = wave >> 1;            // 4 waves along N
    const int hl = lane >> 4;
    const int ln = lane & 15;
    const int mblock = blockIdx.y * 64;
    const int nblock = blockIdx.x * 128;

    f32x8 acc[2][2] = {};

    // A slab: 64 rows x 8 segs = 512 b128 chunks, 2 per thread (branch-free)
    auto issue_a = [&](int k0, int buf) {
#pragma unroll
        for (int i = 0; i < 2; ++i) {
            const int c = tid + i * 256;
            const int row = c >> 3, seg = c & 7;
            async_b128((unsigned)(uintptr_t)&shA[buf][row][seg * 8],
                       A + (size_t)(mblock + row) * lda + k0 + seg * 8);
        }
    };

    // B slab: TDM tile if available, else async fallback (4 b128 per thread)
    auto issue_b = [&](int k0, int buf) {
#if HAVE_TDM
        if (wave == 0)
            tdm_load_2d((unsigned)(uintptr_t)&shB[buf][0][0],
                        Bw + (size_t)nblock * ldb + k0,
                        SLABK, 128, ldb);
#else
#pragma unroll
        for (int i = 0; i < 4; ++i) {
            const int c = tid + i * 256;
            const int row = c >> 3, seg = c & 7;
            async_b128((unsigned)(uintptr_t)&shB[buf][row][seg * 8],
                       Bw + (size_t)(nblock + row) * ldb + k0 + seg * 8);
        }
#endif
    };

    auto wait_copies = [&]() {
        wait_async0();
#if HAVE_TDM
        if (wave == 0) wait_tensor0();
#endif
    };

    auto compute_slab = [&](int buf) {
#pragma unroll
        for (int kk = 0; kk < SLABK / 32; ++kk) {
            Frag32B af[2], bf[2];
#pragma unroll
            for (int mf = 0; mf < 2; ++mf) {
                af[mf].u[0] = *(const uint4*)&shA[buf][wm * 32 + mf * 16 + ln][kk * 32 + hl * 8];
                af[mf].u[1] = *(const uint4*)&shA[buf][wm * 32 + mf * 16 + ln][kk * 32 + hl * 8 + 16];
            }
#pragma unroll
            for (int nf = 0; nf < 2; ++nf) {
                bf[nf].u[0] = *(const uint4*)&shB[buf][wn * 32 + nf * 16 + ln][kk * 32 + hl * 8];
                bf[nf].u[1] = *(const uint4*)&shB[buf][wn * 32 + nf * 16 + ln][kk * 32 + hl * 8 + 16];
            }
#pragma unroll
            for (int mf = 0; mf < 2; ++mf)
#pragma unroll
                for (int nf = 0; nf < 2; ++nf)
                    acc[mf][nf] = wmma_bf16(af[mf], bf[nf], acc[mf][nf]);
        }
    };

    const int nslab = K / SLABK;
    issue_a(0, 0);
    issue_b(0, 0);
    wait_copies();
    __syncthreads();

    for (int ks = 0; ks < nslab; ++ks) {
        const int buf = ks & 1;
        if (ks + 1 < nslab) {          // prefetch next slab into other buffer
            issue_a((ks + 1) * SLABK, buf ^ 1);
            issue_b((ks + 1) * SLABK, buf ^ 1);
        }
        compute_slab(buf);             // overlaps with the in-flight copies
        wait_copies();                 // retire slab ks+1 transfers
        __syncthreads();
    }

    // epilogue: bias + store (direct to global)
#pragma unroll
    for (int nf = 0; nf < 2; ++nf) {
        const int n = nblock + wn * 32 + nf * 16 + ln;
        float bs = 0.0f;
        if (bias0) bs += bias0[n];
        if (bias1) bs += bias1[n];
#pragma unroll
        for (int mf = 0; mf < 2; ++mf) {
#pragma unroll
            for (int r = 0; r < 8; ++r) {
                const int m = mblock + wm * 32 + mf * 16 + r + 8 * hl;
                const size_t row = remapBS
                    ? ((size_t)(m & (B_DIM - 1)) * S_DIM + (size_t)(m >> 5))
                    : (size_t)m;
                C[row * (size_t)ldc + n] = acc[mf][nf][r] + bs;
            }
        }
    }
}

// ---------------------------------------------------------------------------
// One LSTM time step, fused: gates = pre[s] + h_in @ Whh^T ; cell update.
// grid.x = H/32 blocks; block = 8 waves = 4 gates x 2 column-halves.
// Whh (8MB bf16) is L2-resident; direct global fragment loads pipeline fine.
// h double-buffered across steps to avoid cross-block RAW races.
// ---------------------------------------------------------------------------
__global__ __launch_bounds__(256)
void lstm_step_kernel(const float* __restrict__ pre_s,          // [32, 4096]
                      const unsigned short* __restrict__ Whh,   // [4096,1024] bf16
                      const unsigned short* __restrict__ h_in,  // [32,1024] bf16
                      unsigned short* __restrict__ h_out,       // [32,1024] bf16
                      float* __restrict__ c_state,              // [32,1024] f32
                      unsigned short* __restrict__ hs_out) {    // [32,1024] bf16 slice
    __shared__ float lds_g[4][B_DIM][32];

    const int lane = threadIdx.x & 31;
    const int wave = threadIdx.x >> 5;
    const int gate = wave >> 1;                 // 0:i 1:f 2:g 3:o
    const int sub  = wave & 1;                  // which 16 cols of the 32-col chunk
    const int hl = lane >> 4;
    const int ln = lane & 15;
    const int ncol = blockIdx.x * 32 + sub * 16;   // base col in hidden dim
    const int gcol = gate * H_DIM + ncol;          // base col in 4096 gate space

    f32x8 acc0 = {}, acc1 = {};
    const unsigned short* bRow  = Whh  + (size_t)(gcol + ln) * H_DIM + hl * 8;
    const unsigned short* aRow0 = h_in + (size_t)ln         * H_DIM + hl * 8;
    const unsigned short* aRow1 = h_in + (size_t)(16 + ln)  * H_DIM + hl * 8;

    for (int k0 = 0; k0 < H_DIM; k0 += 32) {
        Frag32B bfr;
        bfr.u[0] = *(const uint4*)(bRow + k0);
        bfr.u[1] = *(const uint4*)(bRow + k0 + 16);
        Frag32B a0;
        a0.u[0] = *(const uint4*)(aRow0 + k0);
        a0.u[1] = *(const uint4*)(aRow0 + k0 + 16);
        Frag32B a1;
        a1.u[0] = *(const uint4*)(aRow1 + k0);
        a1.u[1] = *(const uint4*)(aRow1 + k0 + 16);
        acc0 = wmma_bf16(a0, bfr, acc0);
        acc1 = wmma_bf16(a1, bfr, acc1);
    }

    const int nl = sub * 16 + ln;
#pragma unroll
    for (int r = 0; r < 8; ++r) {
        const int b0 = r + 8 * hl;       // rows 0..15
        lds_g[gate][b0][nl]      = acc0[r] + pre_s[(size_t)b0        * G_DIM + gcol + ln];
        lds_g[gate][b0 + 16][nl] = acc1[r] + pre_s[(size_t)(b0 + 16) * G_DIM + gcol + ln];
    }
    __syncthreads();

    for (int idx = threadIdx.x; idx < B_DIM * 32; idx += blockDim.x) {
        const int b = idx >> 5, nc = idx & 31;
        const int n = blockIdx.x * 32 + nc;
        const float gi = sigm(lds_g[0][b][nc]);
        const float gf = sigm(lds_g[1][b][nc]);
        const float gg = tanhf(lds_g[2][b][nc]);
        const float go = sigm(lds_g[3][b][nc]);
        float c = gf * c_state[b * H_DIM + n] + gi * gg;
        c_state[b * H_DIM + n] = c;
        const unsigned short hb = f2bf(go * tanhf(c));
        h_out[b * H_DIM + n] = hb;
        hs_out[(size_t)b * H_DIM + n] = hb;
    }
}

// ---------------------------------------------------------------------------
// Host orchestration
// ---------------------------------------------------------------------------
extern "C" void kernel_launch(void* const* d_in, const int* in_sizes, int n_in,
                              void* d_out, int out_size, void* d_ws, size_t ws_size,
                              hipStream_t stream) {
    const int*   x    = (const int*)  d_in[0];
    const float* embW = (const float*)d_in[1];
    const float* embB = (const float*)d_in[2];
    const float* Wih0 = (const float*)d_in[3];
    const float* Whh0 = (const float*)d_in[4];
    const float* bih0 = (const float*)d_in[5];
    const float* bhh0 = (const float*)d_in[6];
    const float* Wih1 = (const float*)d_in[7];
    const float* Whh1 = (const float*)d_in[8];
    const float* bih1 = (const float*)d_in[9];
    const float* bhh1 = (const float*)d_in[10];
    const float* clsW = (const float*)d_in[11];
    const float* clsB = (const float*)d_in[12];
    float* out = (float*)d_out;

    // Workspace carve (all offsets 256B-aligned)
    char* ws = (char*)d_ws;
    size_t off = 0;
    auto carve = [&](size_t bytes) { char* p = ws + off; off += (bytes + 255) & ~(size_t)255; return p; };

    unsigned short* wih0_bf = (unsigned short*)carve((size_t)G_DIM * EMB_DIM * 2);
    unsigned short* whh0_bf = (unsigned short*)carve((size_t)G_DIM * H_DIM * 2);
    unsigned short* wih1_bf = (unsigned short*)carve((size_t)G_DIM * H_DIM * 2);
    unsigned short* whh1_bf = (unsigned short*)carve((size_t)G_DIM * H_DIM * 2);
    unsigned short* cls_bf  = (unsigned short*)carve((size_t)V_DIM * H_DIM * 2);
    unsigned short* xs_bf   = (unsigned short*)carve((size_t)MROWS * EMB_DIM * 2);
    unsigned short* hs0_bf  = (unsigned short*)carve((size_t)MROWS * H_DIM * 2);
    unsigned short* hs1_bf  = (unsigned short*)carve((size_t)MROWS * H_DIM * 2);
    float*          pre     = (float*)         carve((size_t)MROWS * G_DIM * 4); // reused for both layers
    unsigned short* hA      = (unsigned short*)carve((size_t)B_DIM * H_DIM * 2);
    unsigned short* hB      = (unsigned short*)carve((size_t)B_DIM * H_DIM * 2);
    float*          cstate  = (float*)         carve((size_t)B_DIM * H_DIM * 4);
    (void)ws_size; (void)in_sizes; (void)n_in; (void)out_size;

    // 1) weights -> bf16
    cvt_bf16_kernel<<<2048, 256, 0, stream>>>(Wih0, wih0_bf, (size_t)G_DIM * EMB_DIM);
    cvt_bf16_kernel<<<2048, 256, 0, stream>>>(Whh0, whh0_bf, (size_t)G_DIM * H_DIM);
    cvt_bf16_kernel<<<2048, 256, 0, stream>>>(Wih1, wih1_bf, (size_t)G_DIM * H_DIM);
    cvt_bf16_kernel<<<2048, 256, 0, stream>>>(Whh1, whh1_bf, (size_t)G_DIM * H_DIM);
    cvt_bf16_kernel<<<2048, 256, 0, stream>>>(clsW, cls_bf,  (size_t)V_DIM * H_DIM);

    // 2) embedding gather -> xs [S,B,E] bf16
    embed_kernel<<<MROWS, 256, 0, stream>>>(x, embW, embB, xs_bf);

    // 3) layer-0 input projection: pre = xs @ Wih0^T + bih0 + bhh0
    gemm_bf16_wmma_async<<<dim3(G_DIM / 128, MROWS / 64), 256, 0, stream>>>(
        xs_bf, EMB_DIM, wih0_bf, EMB_DIM, pre, G_DIM, bih0, bhh0,
        MROWS, G_DIM, EMB_DIM, 0);

    // 4) layer-0 recurrent scan
    (void)hipMemsetAsync(hA, 0, (size_t)B_DIM * H_DIM * 2, stream);
    (void)hipMemsetAsync(cstate, 0, (size_t)B_DIM * H_DIM * 4, stream);
    for (int s = 0; s < S_DIM; ++s) {
        const unsigned short* hin = (s & 1) ? hB : hA;
        unsigned short* hout      = (s & 1) ? hA : hB;
        lstm_step_kernel<<<H_DIM / 32, 256, 0, stream>>>(
            pre + (size_t)s * B_DIM * G_DIM, whh0_bf, hin, hout, cstate,
            hs0_bf + (size_t)s * B_DIM * H_DIM);
    }

    // 5) layer-1 input projection: pre = hs0 @ Wih1^T + bih1 + bhh1
    gemm_bf16_wmma_async<<<dim3(G_DIM / 128, MROWS / 64), 256, 0, stream>>>(
        hs0_bf, H_DIM, wih1_bf, H_DIM, pre, G_DIM, bih1, bhh1,
        MROWS, G_DIM, H_DIM, 0);

    // 6) layer-1 recurrent scan
    (void)hipMemsetAsync(hA, 0, (size_t)B_DIM * H_DIM * 2, stream);
    (void)hipMemsetAsync(cstate, 0, (size_t)B_DIM * H_DIM * 4, stream);
    for (int s = 0; s < S_DIM; ++s) {
        const unsigned short* hin = (s & 1) ? hB : hA;
        unsigned short* hout      = (s & 1) ? hA : hB;
        lstm_step_kernel<<<H_DIM / 32, 256, 0, stream>>>(
            pre + (size_t)s * B_DIM * G_DIM, whh1_bf, hin, hout, cstate,
            hs1_bf + (size_t)s * B_DIM * H_DIM);
    }

    // 7) classifier: logits[b,s,:] = hs1[s,b,:] @ cls_W^T + cls_b  (row remap)
    gemm_bf16_wmma_async<<<dim3(V_DIM / 128, MROWS / 64), 256, 0, stream>>>(
        hs1_bf, H_DIM, cls_bf, H_DIM, out, V_DIM, clsB, nullptr,
        MROWS, V_DIM, H_DIM, 1);
}
